// SparseResidualBlock_31868657336338
// MI455X (gfx1250) — matrix operand order repaired and checked by reading, not compiled, not run on hardware
//
#include <hip/hip_runtime.h>

#define NVOX 50000
#define BATCH 4
#define FIN 64
#define FOUT 128
#define TEMBD 256
#define KOFF 27

typedef unsigned short u16;
typedef unsigned char  u8;
typedef unsigned int   u32;
typedef unsigned long long u64;

typedef __attribute__((ext_vector_type(16))) __bf16 v16bf;
typedef __attribute__((ext_vector_type(8)))  float  v8f;
typedef __attribute__((ext_vector_type(4)))  u32    v4u;
typedef __attribute__((ext_vector_type(4)))  float  v4f;
typedef __attribute__((ext_vector_type(4)))  u16    v4s;

union Frag { v16bf v; v4u q[2]; };

// workspace layout (bytes from d_ws; all 256B-aligned)
#define WS_H1ACT 0u          //  6,400,000 B  [N,64]  bf16
#define WS_XB    6400000u    //  6,400,000 B  [N,64]  bf16
#define WS_H2ACT 12800000u   // 12,800,000 B  [N,128] bf16
#define WS_W1T   25600000u   //    442,368 B  [27,128,64]  bf16
#define WS_W2T   26042368u   //    884,736 B  [27,128,128] bf16
#define WS_RESWT 26927104u   //     16,384 B  [128,64]     bf16
#define WS_TEMB  26943488u   //      2,048 B  [4,128]      f32
#define WS_ZPAD  26945536u   //        256 B  zeros (masked-lane redirect)

__device__ __forceinline__ u16 f2bf(float f) {
  u32 u = __float_as_uint(f);
  u32 r = u + 0x7FFFu + ((u >> 16) & 1u);   // round-to-nearest-even
  return (u16)(r >> 16);
}
__device__ __forceinline__ float silu(float x) {
  return x / (1.0f + __expf(-x));
}
__device__ __forceinline__ u32 lds_off(const void* p) {
  // low 32 bits of a generic pointer to LDS == LDS byte offset
  return (u32)(u64)(uintptr_t)p;
}

// ---------------------------------------------------------------------------
// Prep: h1act = bf16(silu(bn1(x))), xb = bf16(x); also zero the redirect pad
// ---------------------------------------------------------------------------
__global__ __launch_bounds__(256) void k_prep_act(
    const float* __restrict__ x,
    const float* __restrict__ g, const float* __restrict__ b,
    const float* __restrict__ mu, const float* __restrict__ var,
    u16* __restrict__ h1act, u16* __restrict__ xb, u32* __restrict__ zbuf) {
  if (blockIdx.x == 0 && threadIdx.x < 64) zbuf[threadIdx.x] = 0;  // 256B zeros
  int i = (blockIdx.x * 256 + threadIdx.x) * 4;
  if (i >= NVOX * FIN) return;
  v4f xv = *(const v4f*)(x + i);
  int c = i & (FIN - 1);
  v4s ha, xa;
#pragma unroll
  for (int j = 0; j < 4; ++j) {
    int cc = c + j;
    float h = (xv[j] - mu[cc]) * rsqrtf(var[cc] + 1e-5f) * g[cc] + b[cc];
    ha[j] = f2bf(silu(h));
    xa[j] = f2bf(xv[j]);
  }
  *(v4s*)(h1act + i) = ha;
  *(v4s*)(xb + i)    = xa;
}

// ---------------------------------------------------------------------------
// FiLM time embedding: temb[b,o] = silu(t[b]) @ t_W + t_b   (4 x 256 x 128)
// ---------------------------------------------------------------------------
__global__ __launch_bounds__(512) void k_temb(
    const float* __restrict__ t, const float* __restrict__ tW,
    const float* __restrict__ tb, float* __restrict__ temb) {
  __shared__ float ts[BATCH * TEMBD];
  int tid = threadIdx.x;
  for (int i = tid; i < BATCH * TEMBD; i += 512) ts[i] = silu(t[i]);
  __syncthreads();
  int b = tid >> 7, o = tid & (FOUT - 1);
  float acc = tb[o];
  for (int c = 0; c < TEMBD; ++c) acc += ts[b * TEMBD + c] * tW[c * FOUT + o];
  temb[b * FOUT + o] = acc;
}

// ---------------------------------------------------------------------------
// Weight transpose + bf16 convert: W[k][fi][fo] (f32) -> Wt[k][fo][fi] (bf16)
// ---------------------------------------------------------------------------
__global__ __launch_bounds__(256) void k_wt(
    const float* __restrict__ W, u16* __restrict__ Wt,
    int Fin, int Fout, int total) {
  int idx = blockIdx.x * 256 + threadIdx.x;
  if (idx >= total) return;
  int fo = idx % Fout;
  int rest = idx / Fout;
  int fi = rest % Fin;
  int k = rest / Fin;
  Wt[((size_t)(k * Fout + fo)) * Fin + fi] = f2bf(W[idx]);
}

// ---------------------------------------------------------------------------
// conv1 (gather-GEMM, 64->128) fused with +temb, BN2, SiLU -> bf16 h2act
// 256 threads = 8 waves; tile = 16 voxels; wave w owns FOUT cols [16w,16w+16)
// Gather: async global->LDS (b64/lane, GVS mode), double-buffered, one
// barrier per tap. All 27 per-lane byte offsets precomputed in registers;
// masked taps redirect branchlessly to a zeroed pad (same SGPR base).
// ---------------------------------------------------------------------------
__global__ __launch_bounds__(256) void k_conv1(
    const u16* __restrict__ h1act, const int* __restrict__ nbr,
    const u8* __restrict__ mask, const u16* __restrict__ W1t,
    const float* __restrict__ temb, const int* __restrict__ batch_idx,
    const float* __restrict__ g2, const float* __restrict__ b2,
    const float* __restrict__ mu2, const float* __restrict__ var2,
    u32 zoff, u16* __restrict__ h2act) {
  __shared__ __align__(16) u16 As[2][16 * FIN];

  const int tile = blockIdx.x;
  const int tid  = threadIdx.x;
  const int lane = tid & 31;
  const int wave = tid >> 5;
  const int grow = tid >> 4;          // gather row 0..15
  const int gcol = (tid & 15) * 4;    // 4 bf16 (8B) per thread
  const int vg   = tile * 16 + grow;
  const int ncol = (wave << 4) + (lane & 15);
  const int hi   = lane >> 4;

  const u64 base = (u64)(uintptr_t)h1act;

  // precompute all gather offsets (branchless mask fold)
  u32 offs[KOFF];
#pragma unroll
  for (int k = 0; k < KOFF; ++k) {
    int nb = nbr[k * NVOX + vg];
    int m  = mask[k * NVOX + vg];
    u32 o  = (u32)nb * (FIN * 2) + gcol * 2;
    offs[k] = m ? o : zoff;
  }

  const u32 dst0 = lds_off(&As[0][grow * FIN + gcol]);
  const u32 dst1 = lds_off(&As[1][grow * FIN + gcol]);

  asm volatile("global_load_async_to_lds_b64 %0, %1, %2"
               :: "v"(dst0), "v"(offs[0]), "s"(base) : "memory");
  v8f acc = {};
#pragma unroll
  for (int k = 0; k < KOFF; ++k) {
    const int buf = k & 1;
    asm volatile("s_wait_asynccnt 0x0" ::: "memory");  // tile k landed in LDS
    __syncthreads();                                   // visible to all waves;
                                                       // buf^1 free to overwrite
    if (k + 1 < KOFF)
      asm volatile("global_load_async_to_lds_b64 %0, %1, %2"
                   :: "v"(buf ? dst0 : dst1), "v"(offs[k + 1]), "s"(base)
                   : "memory");
#pragma unroll
    for (int kk = 0; kk < 2; ++kk) {            // K = 64 -> two K=32 steps
      const int kbase = kk * 32;
      Frag a, bf;
      // A 16x32 bf16: lane<16: K [0..7],[16..23]; lane>=16: K [8..15],[24..31]
      const int aoff = (lane & 15) * FIN + kbase + hi * 8;
      a.q[0] = *(const v4u*)(&As[buf][aoff]);
      a.q[1] = *(const v4u*)(&As[buf][aoff + 16]);
      // B 32x16 bf16: lane<16: K [0..7],[16..23]; lane>=16: K [8..15],[24..31]
      const u16* bp = W1t + ((size_t)(k * FOUT + ncol)) * FIN + kbase + hi * 8;
      bf.q[0] = *(const v4u*)(bp);
      bf.q[1] = *(const v4u*)(bp + 16);
      acc = __builtin_amdgcn_wmma_f32_16x16x32_bf16(
          false, a.v, false, bf.v, (short)0, acc, false, false);
    }
  }
  const float gg = g2[ncol], bb = b2[ncol], mm = mu2[ncol];
  const float iv = rsqrtf(var2[ncol] + 1e-5f);
  const int mbase = hi * 8;
#pragma unroll
  for (int r = 0; r < 8; ++r) {
    const int vv = tile * 16 + mbase + r;
    float val = acc[r] + temb[batch_idx[vv] * FOUT + ncol];
    float hn = (val - mm) * iv * gg + bb;
    h2act[(size_t)vv * FOUT + ncol] = f2bf(silu(hn));
  }
}

// ---------------------------------------------------------------------------
// conv2 (gather-GEMM, 128->128) + residual GEMM (xb @ res_W) + res_b -> f32
// ---------------------------------------------------------------------------
__global__ __launch_bounds__(256) void k_conv2(
    const u16* __restrict__ h2act, const int* __restrict__ nbr,
    const u8* __restrict__ mask, const u16* __restrict__ W2t,
    const u16* __restrict__ xb, const u16* __restrict__ resWt,
    const float* __restrict__ res_b, u32 zoff, float* __restrict__ out) {
  __shared__ __align__(16) u16 As[2][16 * FOUT];

  const int tile = blockIdx.x;
  const int tid  = threadIdx.x;
  const int lane = tid & 31;
  const int wave = tid >> 5;
  const int grow = tid >> 4;
  const int gcol = (tid & 15) * 8;    // 8 bf16 (16B) per thread
  const int vg   = tile * 16 + grow;
  const int ncol = (wave << 4) + (lane & 15);
  const int hi   = lane >> 4;

  const u64 base = (u64)(uintptr_t)h2act;

  u32 offs[KOFF];
#pragma unroll
  for (int k = 0; k < KOFF; ++k) {
    int nb = nbr[k * NVOX + vg];
    int m  = mask[k * NVOX + vg];
    u32 o  = (u32)nb * (FOUT * 2) + gcol * 2;
    offs[k] = m ? o : zoff;
  }

  const u32 dst0 = lds_off(&As[0][grow * FOUT + gcol]);
  const u32 dst1 = lds_off(&As[1][grow * FOUT + gcol]);

  asm volatile("global_load_async_to_lds_b128 %0, %1, %2"
               :: "v"(dst0), "v"(offs[0]), "s"(base) : "memory");
  v8f acc = {};
#pragma unroll
  for (int k = 0; k < KOFF; ++k) {
    const int buf = k & 1;
    asm volatile("s_wait_asynccnt 0x0" ::: "memory");
    __syncthreads();
    if (k + 1 < KOFF)
      asm volatile("global_load_async_to_lds_b128 %0, %1, %2"
                   :: "v"(buf ? dst0 : dst1), "v"(offs[k + 1]), "s"(base)
                   : "memory");
#pragma unroll
    for (int kk = 0; kk < 4; ++kk) {            // K = 128 -> four K=32 steps
      const int kbase = kk * 32;
      Frag a, bf;
      const int aoff = (lane & 15) * FOUT + kbase + hi * 8;
      a.q[0] = *(const v4u*)(&As[buf][aoff]);
      a.q[1] = *(const v4u*)(&As[buf][aoff + 16]);
      const u16* bp = W2t + ((size_t)(k * FOUT + ncol)) * FOUT + kbase + hi * 8;
      bf.q[0] = *(const v4u*)(bp);
      bf.q[1] = *(const v4u*)(bp + 16);
      acc = __builtin_amdgcn_wmma_f32_16x16x32_bf16(
          false, a.v, false, bf.v, (short)0, acc, false, false);
    }
  }
  // residual: identity row map, A fragments straight from global xb
#pragma unroll
  for (int kk = 0; kk < 2; ++kk) {
    const int kbase = kk * 32;
    Frag a, bf;
    const u16* ap = xb + (size_t)(tile * 16 + (lane & 15)) * FIN + kbase + hi * 8;
    a.q[0] = *(const v4u*)(ap);
    a.q[1] = *(const v4u*)(ap + 16);
    const u16* bp = resWt + (size_t)ncol * FIN + kbase + hi * 8;
    bf.q[0] = *(const v4u*)(bp);
    bf.q[1] = *(const v4u*)(bp + 16);
    acc = __builtin_amdgcn_wmma_f32_16x16x32_bf16(
        false, a.v, false, bf.v, (short)0, acc, false, false);
  }
  const float rb = res_b[ncol];
  const int mbase = hi * 8;
#pragma unroll
  for (int r = 0; r < 8; ++r) {
    const int vv = tile * 16 + mbase + r;
    out[(size_t)vv * FOUT + ncol] = acc[r] + rb;
  }
}

// ---------------------------------------------------------------------------
extern "C" void kernel_launch(void* const* d_in, const int* in_sizes, int n_in,
                              void* d_out, int out_size, void* d_ws, size_t ws_size,
                              hipStream_t stream) {
  const float* x_feats   = (const float*)d_in[0];
  const int*   batch_idx = (const int*)d_in[1];
  const float* t         = (const float*)d_in[2];
  const int*   nbr1      = (const int*)d_in[3];
  const u8*    mask1     = (const u8*)d_in[4];
  const int*   nbr2      = (const int*)d_in[5];
  const u8*    mask2     = (const u8*)d_in[6];
  const float* bn1_g     = (const float*)d_in[7];
  const float* bn1_b     = (const float*)d_in[8];
  const float* bn1_m     = (const float*)d_in[9];
  const float* bn1_v     = (const float*)d_in[10];
  const float* W1        = (const float*)d_in[11];
  const float* t_W       = (const float*)d_in[12];
  const float* t_b       = (const float*)d_in[13];
  const float* bn2_g     = (const float*)d_in[14];
  const float* bn2_b     = (const float*)d_in[15];
  const float* bn2_m     = (const float*)d_in[16];
  const float* bn2_v     = (const float*)d_in[17];
  const float* W2        = (const float*)d_in[18];
  const float* res_W     = (const float*)d_in[19];
  const float* res_b     = (const float*)d_in[20];

  char* ws = (char*)d_ws;
  u16*   h1act = (u16*)(ws + WS_H1ACT);
  u16*   xb    = (u16*)(ws + WS_XB);
  u16*   h2act = (u16*)(ws + WS_H2ACT);
  u16*   W1t   = (u16*)(ws + WS_W1T);
  u16*   W2t   = (u16*)(ws + WS_W2T);
  u16*   resWt = (u16*)(ws + WS_RESWT);
  float* temb  = (float*)(ws + WS_TEMB);
  u32*   zbuf  = (u32*)(ws + WS_ZPAD);

  const int nTiles = NVOX / 16;                  // 3125, exact

  k_prep_act<<<(NVOX * FIN / 4 + 255) / 256, 256, 0, stream>>>(
      x_feats, bn1_g, bn1_b, bn1_m, bn1_v, h1act, xb, zbuf);

  k_temb<<<1, 512, 0, stream>>>(t, t_W, t_b, temb);

  k_wt<<<(KOFF * FIN * FOUT + 255) / 256, 256, 0, stream>>>(
      W1, W1t, FIN, FOUT, KOFF * FIN * FOUT);
  k_wt<<<(KOFF * FOUT * FOUT + 255) / 256, 256, 0, stream>>>(
      W2, W2t, FOUT, FOUT, KOFF * FOUT * FOUT);
  k_wt<<<(FIN * FOUT + 255) / 256, 256, 0, stream>>>(
      res_W, resWt, FIN, FOUT, FIN * FOUT);

  // zero-pad offsets relative to each conv's gather base
  const u32 zoff1 = WS_ZPAD - WS_H1ACT;
  const u32 zoff2 = WS_ZPAD - WS_H2ACT;

  k_conv1<<<nTiles, 256, 0, stream>>>(
      h1act, nbr1, mask1, W1t, temb, batch_idx,
      bn2_g, bn2_b, bn2_m, bn2_v, zoff1, h2act);

  k_conv2<<<nTiles, 256, 0, stream>>>(
      h2act, nbr2, mask2, W2t, xb, resWt, res_b, zoff2, (float*)d_out);
}
